// CorrelationLayer_10797547782551
// MI455X (gfx1250) — matrix operand order repaired and checked by reading, not compiled, not run on hardware
//
#include <hip/hip_runtime.h>
#include <hip/hip_bf16.h>
#include <stdint.h>

// ---- problem constants (from reference) ----
#define N_B     4
#define C_CH    64
#define H_IMG   128
#define W_IMG   128
#define PAD     20
#define D_DISP  41                  // 2*PAD+1
#define DD      (D_DISP * D_DISP)   // 1681
#define HP      (H_IMG + 2 * PAD)   // 168 padded rows
#define WP      184                 // 168 padded cols + 16 zero guard (dx0=48 tile)

typedef __attribute__((ext_vector_type(16))) __bf16 v16bf;
typedef __attribute__((ext_vector_type(8)))  float  v8f;

static __device__ __forceinline__ unsigned short f2bf(float f) {
  unsigned u = __float_as_uint(f);
  u += 0x7FFFu + ((u >> 16) & 1u);   // round-to-nearest-even
  return (unsigned short)(u >> 16);
}

// ---- pack fm1: NCHW f32 -> NHWC bf16 (channels contiguous per pixel) ----
__global__ void __launch_bounds__(256)
pack_fm1(const float* __restrict__ src, unsigned short* __restrict__ dst) {
  const int tid = blockIdx.x * blockDim.x + threadIdx.x;
  const int total = N_B * H_IMG * W_IMG * 8;         // 8 chunks of 8 channels
  if (tid >= total) return;
  const int chunk = tid & 7;
  const int pixel = tid >> 3;
  const int x = pixel % W_IMG;
  const int y = (pixel / W_IMG) % H_IMG;
  const int n = pixel / (W_IMG * H_IMG);
  const int c0 = chunk * 8;
  const size_t plane = (size_t)H_IMG * W_IMG;
  const size_t sbase = ((size_t)n * C_CH + c0) * plane + (size_t)y * W_IMG + x;
  unsigned short t[8];
#pragma unroll
  for (int j = 0; j < 8; ++j) t[j] = f2bf(src[sbase + (size_t)j * plane]);
  uint4 o;
  o.x = (unsigned)t[0] | ((unsigned)t[1] << 16);
  o.y = (unsigned)t[2] | ((unsigned)t[3] << 16);
  o.z = (unsigned)t[4] | ((unsigned)t[5] << 16);
  o.w = (unsigned)t[6] | ((unsigned)t[7] << 16);
  *(uint4*)(dst + (size_t)pixel * C_CH + c0) = o;    // 16B coalesced store
}

// ---- pack fm2: NCHW f32 -> padded NHWC bf16 [n][168][184][64], zeros outside ----
__global__ void __launch_bounds__(256)
pack_fm2(const float* __restrict__ src, unsigned short* __restrict__ dst) {
  const int tid = blockIdx.x * blockDim.x + threadIdx.x;
  const int total = N_B * HP * WP * 8;
  if (tid >= total) return;
  const int chunk = tid & 7;
  const int pixel = tid >> 3;
  const int xp = pixel % WP;
  const int yp = (pixel / WP) % HP;
  const int n  = pixel / (WP * HP);
  const int x = xp - PAD, y = yp - PAD;
  const bool inb = (x >= 0) & (x < W_IMG) & (y >= 0) & (y < H_IMG);
  const int c0 = chunk * 8;
  const size_t plane = (size_t)H_IMG * W_IMG;
  const size_t sbase = ((size_t)n * C_CH + c0) * plane + (size_t)y * W_IMG + x;
  unsigned short t[8];
#pragma unroll
  for (int j = 0; j < 8; ++j) t[j] = inb ? f2bf(src[sbase + (size_t)j * plane]) : (unsigned short)0;
  uint4 o;
  o.x = (unsigned)t[0] | ((unsigned)t[1] << 16);
  o.y = (unsigned)t[2] | ((unsigned)t[3] << 16);
  o.z = (unsigned)t[4] | ((unsigned)t[5] << 16);
  o.w = (unsigned)t[6] | ((unsigned)t[7] << 16);
  *(uint4*)(dst + (size_t)pixel * C_CH + c0) = o;
}

// ---- main correlation via banded bf16 WMMA ----
// block = (x-tile of 16 pixels, row y, batch n); 8 waves stride dy = 0..40.
// Per (dy, dx0): D[m,n] = sum_c A[m,c]*B[n,c] -> displacement d = dx0 + n - m.
__global__ void __launch_bounds__(256)
corr_wmma(const unsigned short* __restrict__ f1t,
          const unsigned short* __restrict__ f2p,
          float* __restrict__ out) {
  const int xt   = blockIdx.x;          // 0..7
  const int y    = blockIdx.y;          // 0..127
  const int n    = blockIdx.z;          // 0..3
  const int lane = threadIdx.x & 31;
  const int wave = threadIdx.x >> 5;
  const int x0   = xt * 16;
  const int mrow = lane & 15;
  const int half = lane >> 4;

  // A fragments (16 pixels x 64 ch), contiguous K layout (matches B):
  // half 0: K0..15 ; half 1: K16..31
  const unsigned short* a_ptr =
      f1t + (((size_t)(n * H_IMG + y) * W_IMG) + x0 + mrow) * C_CH;
  union { uint4 q[2]; v16bf v; } a_lo, a_hi;
  a_lo.q[0] = *(const uint4*)(a_ptr + half * 16);
  a_lo.q[1] = *(const uint4*)(a_ptr + half * 16 + 8);
  a_hi.q[0] = *(const uint4*)(a_ptr + 32 + half * 16);
  a_hi.q[1] = *(const uint4*)(a_ptr + 32 + half * 16 + 8);

  const size_t out_base = ((size_t)(n * H_IMG) + y) * W_IMG * DD;
  const int ncol = lane & 15;

  for (int dy = wave; dy < D_DISP; dy += 8) {
    const unsigned short* brow =
        f2p + ((size_t)(n * HP + (y + dy)) * WP) * C_CH;
    for (int dx0 = 0; dx0 < 64; dx0 += 16) {
      // B fragments: lane holds column n = lane&15, K-run = half*16..half*16+15
      const unsigned short* b_ptr = brow + (size_t)(x0 + dx0 + ncol) * C_CH;
      const int kb = half * 16;
      union { uint4 q[2]; v16bf v; } b_lo, b_hi;
      b_lo.q[0] = *(const uint4*)(b_ptr + kb);
      b_lo.q[1] = *(const uint4*)(b_ptr + kb + 8);
      b_hi.q[0] = *(const uint4*)(b_ptr + 32 + kb);
      b_hi.q[1] = *(const uint4*)(b_ptr + 32 + kb + 8);

      v8f acc = {0.f, 0.f, 0.f, 0.f, 0.f, 0.f, 0.f, 0.f};
      acc = __builtin_amdgcn_wmma_f32_16x16x32_bf16(
          false, a_lo.v, false, b_lo.v, (short)0, acc, false, false);
      acc = __builtin_amdgcn_wmma_f32_16x16x32_bf16(
          false, a_hi.v, false, b_hi.v, (short)0, acc, false, false);

      // D layout: VGPR v, lanes 0-15 -> M=v, lanes 16-31 -> M=v+8; N = lane&15
#pragma unroll
      for (int v = 0; v < 8; ++v) {
        const int ms = v + half * 8;
        const int d  = dx0 + ncol - ms;               // displacement index
        if ((unsigned)d < (unsigned)D_DISP) {
          out[out_base + (size_t)(x0 + ms) * DD + (size_t)dy * D_DISP + d] = acc[v];
        }
      }
    }
  }
}

extern "C" void kernel_launch(void* const* d_in, const int* in_sizes, int n_in,
                              void* d_out, int out_size, void* d_ws, size_t ws_size,
                              hipStream_t stream) {
  (void)in_sizes; (void)n_in; (void)out_size; (void)ws_size;
  const float* fm1 = (const float*)d_in[0];
  const float* fm2 = (const float*)d_in[1];
  float* out = (float*)d_out;

  unsigned short* f1t = (unsigned short*)d_ws;                       // 8.39 MB
  unsigned short* f2p = f1t + (size_t)N_B * H_IMG * W_IMG * C_CH;    // 15.83 MB
  // total workspace: 24.22 MB

  {
    const int total = N_B * H_IMG * W_IMG * 8;
    pack_fm1<<<(total + 255) / 256, 256, 0, stream>>>(fm1, f1t);
  }
  {
    const int total = N_B * HP * WP * 8;
    pack_fm2<<<(total + 255) / 256, 256, 0, stream>>>(fm2, f2p);
  }
  {
    dim3 grid(W_IMG / 16, H_IMG, N_B);   // 8 x 128 x 4 blocks, 256 thr (8 waves)
    corr_wmma<<<grid, 256, 0, stream>>>(f1t, f2p, out);
  }
}
